// TemporalBlock_Attention_64690797412357
// MI455X (gfx1250) — matrix operand order repaired and checked by reading, not compiled, not run on hardware
//
#include <hip/hip_runtime.h>
#include <hip/hip_bf16.h>
#include <math.h>

// ---------------- problem constants ----------------
#define BSZ   8
#define CCH   256
#define LSEQ  4096
#define KTAP  7
#define DILC  4
#define PADL  24          // DIL*(K-1)
#define LP    4120        // LSEQ + PADL
#define NHEAD 4
#define DH    64
#define GN    1030        // LP / DILC
#define NQKV  768
#define MROWS (BSZ * LP)  // 32960

typedef __attribute__((ext_vector_type(16))) _Float16 v16h;
typedef __attribute__((ext_vector_type(8)))  _Float16 v8h;
typedef __attribute__((ext_vector_type(8)))  float    v8f;

// ---------------- WMMA fragment helpers (ISA 7.12.2 layouts, wave32) ----------
// A fragment 16x32 f16, source row-major [M][K], row stride lda (elements).
// lanes 0-15 : row = lane,    K {0..7} in v[0..7], K {16..23} in v[8..15]
// lanes 16-31: row = lane-16, K {8..15},          K {24..31}
__device__ __forceinline__ v16h load_frag_a(const _Float16* __restrict__ base,
                                            int lda, int lane) {
  const int row  = lane & 15;
  const int koff = (lane >> 4) * 8;
  const _Float16* p = base + (size_t)row * lda + koff;
  union { v16h v; v8h h[2]; } u;
  u.h[0] = *(const v8h*)(p);
  u.h[1] = *(const v8h*)(p + 16);
  return u.v;
}

// B fragment 32x16 f16, source stored N-major [N][K], row stride ldb.
// lanes 0-15 : col = lane,    K {0..7, 16..23}; lanes 16-31: col = lane-16, K {8..15, 24..31}
__device__ __forceinline__ v16h load_frag_b(const _Float16* __restrict__ base,
                                            int ldb, int lane) {
  const int row  = lane & 15;
  const int koff = (lane >> 4) * 8;
  const _Float16* p = base + (size_t)row * ldb + koff;
  union { v16h v; v8h h[2]; } u;
  u.h[0] = *(const v8h*)(p);
  u.h[1] = *(const v8h*)(p + 16);
  return u.v;
}

__device__ __forceinline__ v8f wmma16(v16h a, v16h b, v8f c) {
  return __builtin_amdgcn_wmma_f32_16x16x32_f16(false, a, false, b,
                                                (short)0, c, false, false);
}

__device__ __forceinline__ float elu1(float x) {
  return x > 0.f ? x : (__expf(x) - 1.f);
}

// ---------------- kernel 1: weight-norm conv weights -> f16 [tap][co][ci] ----
__global__ __launch_bounds__(256)
void k_conv_weight(const float* __restrict__ conv_v, const float* __restrict__ conv_g,
                   _Float16* __restrict__ wconv_t) {
  const int co = blockIdx.x;            // 256 blocks
  const int t  = threadIdx.x;
  __shared__ float red[256];
  float s = 0.f;
  for (int idx = t; idx < CCH * KTAP; idx += 256) {
    float w = conv_v[(size_t)co * CCH * KTAP + idx];
    s += w * w;
  }
  red[t] = s;
  __syncthreads();
  for (int off = 128; off > 0; off >>= 1) {
    if (t < off) red[t] += red[t + off];
    __syncthreads();
  }
  const float scale = conv_g[co] * rsqrtf(red[0]);
  for (int idx = t; idx < CCH * KTAP; idx += 256) {
    int ci = idx / KTAP, k = idx % KTAP;
    wconv_t[((size_t)k * CCH + co) * CCH + ci] =
        (_Float16)(conv_v[(size_t)co * CCH * KTAP + idx] * scale);
  }
}

// ---------------- kernel 2: transpose fp32 [rows][cols] -> f16 [cols][rows] --
__global__ __launch_bounds__(256)
void k_transpose_w(const float* __restrict__ w, _Float16* __restrict__ wt,
                   int rows, int cols) {
  int idx = blockIdx.x * 256 + threadIdx.x;
  if (idx >= rows * cols) return;
  int n = idx / rows, c = idx % rows;
  wt[(size_t)n * rows + c] = (_Float16)w[(size_t)c * cols + n];
}

// ---------------- kernel 3: x (B,C,L) fp32 -> xT (B,LP,C) f16, 24 zero rows --
__global__ __launch_bounds__(256)
void k_transpose_x(const float* __restrict__ x, _Float16* __restrict__ xT) {
  __shared__ _Float16 tile[32][33];
  const int b  = blockIdx.z;
  const int p0 = blockIdx.x * 32;
  const int c0 = blockIdx.y * 32;
  const int tx = threadIdx.x & 31;
  const int ty = threadIdx.x >> 5;    // 0..7
#pragma unroll
  for (int r = 0; r < 4; ++r) {
    int ci = c0 + ty + 8 * r;
    int l  = p0 + tx - PADL;
    float v = 0.f;
    if (l >= 0 && l < LSEQ) v = x[((size_t)b * CCH + ci) * LSEQ + l];
    tile[ty + 8 * r][tx] = (_Float16)v;
  }
  __syncthreads();
#pragma unroll
  for (int r = 0; r < 4; ++r) {
    int p = p0 + ty + 8 * r;
    if (p < LP)
      xT[((size_t)b * LP + p) * CCH + c0 + tx] = tile[tx][ty + 8 * r];
  }
}

// ---------------- kernel 4: zero the 24 left-pad rows of xs -----------------
__global__ __launch_bounds__(256)
void k_zero_pad(_Float16* __restrict__ xs) {
  int idx = blockIdx.x * 256 + threadIdx.x;  // B*24*256 threads
  int b   = idx / (PADL * CCH);
  int rem = idx % (PADL * CCH);
  xs[(size_t)b * LP * CCH + rem] = (_Float16)0.f;
}

// ---------------- kernel 5: dilated conv as implicit WMMA GEMM --------------
// out[b,l,co] = sum_t sum_ci xT[b, l+4t, ci] * wconv_t[t, co, ci]; +bias, ELU
// grid.x = B * (LSEQ/64), grid.y = CCH/128; 128 thr = 4 waves, 32x64 per wave
__global__ __launch_bounds__(128)
void k_conv_gemm(const _Float16* __restrict__ xT, const _Float16* __restrict__ wconv_t,
                 const float* __restrict__ conv_b, _Float16* __restrict__ xs) {
  const int lane = threadIdx.x & 31;
  const int wave = threadIdx.x >> 5;
  const int wm = wave >> 1, wn = wave & 1;
  const int b     = blockIdx.x >> 6;
  const int lbase = (blockIdx.x & 63) * 64 + wm * 32;
  const int nbase = blockIdx.y * 128 + wn * 64;
  const _Float16* xb = xT + (size_t)b * LP * CCH;
  v8f acc[2][4] = {};
  for (int t = 0; t < KTAP; ++t) {
    const _Float16* arow = xb + (size_t)(lbase + DILC * t) * CCH;
    const _Float16* wt   = wconv_t + (size_t)t * CCH * CCH;
#pragma unroll
    for (int k = 0; k < CCH; k += 32) {
      v16h a0 = load_frag_a(arow + k, CCH, lane);
      v16h a1 = load_frag_a(arow + (size_t)16 * CCH + k, CCH, lane);
#pragma unroll
      for (int j = 0; j < 4; ++j) {
        v16h bf = load_frag_b(wt + (size_t)(nbase + 16 * j) * CCH + k, CCH, lane);
        acc[0][j] = wmma16(a0, bf, acc[0][j]);
        acc[1][j] = wmma16(a1, bf, acc[1][j]);
      }
    }
  }
#pragma unroll
  for (int i = 0; i < 2; ++i)
#pragma unroll
    for (int j = 0; j < 4; ++j)
#pragma unroll
      for (int r = 0; r < 8; ++r) {
        int l  = lbase + i * 16 + (lane >> 4) * 8 + r;
        int co = nbase + j * 16 + (lane & 15);
        float v = elu1(acc[i][j][r] + conv_b[co]);
        xs[((size_t)b * LP + l + PADL) * CCH + co] = (_Float16)v;
      }
}

// ---------------- kernel 6: QKV WMMA GEMM + scatter -------------------------
// (B*LP, 256) x (256, 768): A = xs row-major, B = wqkv_t [N][K]
__global__ __launch_bounds__(128)
void k_qkv_gemm(const _Float16* __restrict__ xs, const _Float16* __restrict__ wqkv_t,
                const float* __restrict__ bqkv, _Float16* __restrict__ qarr,
                _Float16* __restrict__ karr, _Float16* __restrict__ varr) {
  const int lane = threadIdx.x & 31;
  const int wave = threadIdx.x >> 5;
  const int wm = wave >> 1, wn = wave & 1;
  const int tileM = blockIdx.x * 64 + wm * 32;
  const int tileN = blockIdx.y * 128 + wn * 64;
  v8f acc[2][4] = {};
#pragma unroll
  for (int k = 0; k < CCH; k += 32) {
    v16h a0 = load_frag_a(xs + (size_t)tileM * CCH + k, CCH, lane);
    v16h a1 = load_frag_a(xs + (size_t)(tileM + 16) * CCH + k, CCH, lane);
#pragma unroll
    for (int j = 0; j < 4; ++j) {
      v16h bf = load_frag_b(wqkv_t + (size_t)(tileN + 16 * j) * CCH + k, CCH, lane);
      acc[0][j] = wmma16(a0, bf, acc[0][j]);
      acc[1][j] = wmma16(a1, bf, acc[1][j]);
    }
  }
#pragma unroll
  for (int i = 0; i < 2; ++i)
#pragma unroll
    for (int j = 0; j < 4; ++j)
#pragma unroll
      for (int r = 0; r < 8; ++r) {
        int m = tileM + i * 16 + (lane >> 4) * 8 + r;
        int n = tileN + j * 16 + (lane & 15);
        float v = acc[i][j][r] + bqkv[n];
        int b = m / LP, p = m % LP;
        int g = p >> 2, d = p & 3;
        int which = n >> 8, h = (n >> 6) & 3, e = n & 63;
        size_t dst = ((((size_t)b * DILC + d) * NHEAD + h) * GN + g) * DH + e;
        if (which == 0)      qarr[dst] = (_Float16)(v * 0.125f);  // Dh^-0.5
        else if (which == 1) karr[dst] = (_Float16)v;
        else                 varr[dst] = (_Float16)v;
      }
}

// ---------------- kernel 7: neighborhood attention (VALU) -------------------
__global__ __launch_bounds__(256)
void k_attn(const _Float16* __restrict__ qarr, const _Float16* __restrict__ karr,
            const _Float16* __restrict__ varr, const float* __restrict__ rpb,
            _Float16* __restrict__ oarr) {
  const int s = blockIdx.x;                       // ((b*4+d)*4+h), 0..127
  const int g = blockIdx.y * 256 + threadIdx.x;
  if (g >= GN) return;
  const int h = s & 3, d = (s >> 2) & 3, b = s >> 4;
  const _Float16* qp    = qarr + ((size_t)s * GN + g) * DH;
  const _Float16* kbase = karr + (size_t)s * GN * DH;
  const _Float16* vbase = varr + (size_t)s * GN * DH;
  v8h qv[8];
#pragma unroll
  for (int c = 0; c < 8; ++c) qv[c] = *(const v8h*)(qp + 8 * c);
  int start = g - (KTAP / 2);
  if (start < 0) start = 0;
  if (start > GN - KTAP) start = GN - KTAP;
  float logits[KTAP];
#pragma unroll
  for (int j = 0; j < KTAP; ++j) {
    const _Float16* kp = kbase + (size_t)(start + j) * DH;
    float dot = 0.f;
#pragma unroll
    for (int c = 0; c < 8; ++c) {
      v8h kc = *(const v8h*)(kp + 8 * c);
#pragma unroll
      for (int u = 0; u < 8; ++u) dot += (float)qv[c][u] * (float)kc[u];
    }
    logits[j] = dot + rpb[h * (2 * KTAP - 1) + (start + j - g + KTAP - 1)];
  }
  float mx = logits[0];
#pragma unroll
  for (int j = 1; j < KTAP; ++j) mx = fmaxf(mx, logits[j]);
  float wsum = 0.f;
#pragma unroll
  for (int j = 0; j < KTAP; ++j) { logits[j] = __expf(logits[j] - mx); wsum += logits[j]; }
  const float inv = 1.f / wsum;
  float o[DH];
#pragma unroll
  for (int e = 0; e < DH; ++e) o[e] = 0.f;
#pragma unroll
  for (int j = 0; j < KTAP; ++j) {
    const float wj = logits[j] * inv;
    const _Float16* vp = vbase + (size_t)(start + j) * DH;
#pragma unroll
    for (int c = 0; c < 8; ++c) {
      v8h vc = *(const v8h*)(vp + 8 * c);
#pragma unroll
      for (int u = 0; u < 8; ++u) o[8 * c + u] += wj * (float)vc[u];
    }
  }
  const int p = 4 * g + d;
  _Float16* op = oarr + ((size_t)b * LP + p) * CCH + h * DH;
#pragma unroll
  for (int e = 0; e < DH; ++e) op[e] = (_Float16)o[e];
}

// ---------------- kernel 8: proj WMMA GEMM + crop/ELU/residual --------------
__global__ __launch_bounds__(128)
void k_proj_gemm(const _Float16* __restrict__ oarr, const _Float16* __restrict__ wproj_t,
                 const float* __restrict__ bproj, const float* __restrict__ x,
                 float* __restrict__ dres, float* __restrict__ dout) {
  const int lane = threadIdx.x & 31;
  const int wave = threadIdx.x >> 5;
  const int wm = wave >> 1, wn = wave & 1;
  const int tileM = blockIdx.x * 64 + wm * 32;
  const int tileN = blockIdx.y * 128 + wn * 64;
  v8f acc[2][4] = {};
#pragma unroll
  for (int k = 0; k < CCH; k += 32) {
    v16h a0 = load_frag_a(oarr + (size_t)tileM * CCH + k, CCH, lane);
    v16h a1 = load_frag_a(oarr + (size_t)(tileM + 16) * CCH + k, CCH, lane);
#pragma unroll
    for (int j = 0; j < 4; ++j) {
      v16h bf = load_frag_b(wproj_t + (size_t)(tileN + 16 * j) * CCH + k, CCH, lane);
      acc[0][j] = wmma16(a0, bf, acc[0][j]);
      acc[1][j] = wmma16(a1, bf, acc[1][j]);
    }
  }
#pragma unroll
  for (int i = 0; i < 2; ++i)
#pragma unroll
    for (int j = 0; j < 4; ++j)
#pragma unroll
      for (int r = 0; r < 8; ++r) {
        int m = tileM + i * 16 + (lane >> 4) * 8 + r;
        int n = tileN + j * 16 + (lane & 15);
        float v = acc[i][j][r] + bproj[n];
        int b = m / LP, p = m % LP;
        if (p < LSEQ) {                      // final crop keeps p in [0,4096)
          float ev = elu1(v);
          size_t oidx = ((size_t)b * CCH + n) * LSEQ + p;
          dres[oidx] = ev + x[oidx];
          dout[oidx] = ev;
        }
      }
}

// ---------------- host-side launch ------------------------------------------
extern "C" void kernel_launch(void* const* d_in, const int* in_sizes, int n_in,
                              void* d_out, int out_size, void* d_ws, size_t ws_size,
                              hipStream_t stream) {
  const float* x      = (const float*)d_in[0];
  const float* conv_v = (const float*)d_in[1];
  const float* conv_g = (const float*)d_in[2];
  const float* conv_b = (const float*)d_in[3];
  const float* wqkv   = (const float*)d_in[4];
  const float* bqkv   = (const float*)d_in[5];
  const float* rpb    = (const float*)d_in[6];
  const float* wproj  = (const float*)d_in[7];
  const float* bproj  = (const float*)d_in[8];

  // workspace carve-out (f16 buffers), 256B aligned
  char* ws = (char*)d_ws;
  size_t off = 0;
  auto alloc = [&](size_t bytes) {
    char* p = ws + off;
    off = (off + bytes + 255) & ~(size_t)255;
    return p;
  };
  _Float16* wconv_t = (_Float16*)alloc((size_t)KTAP * CCH * CCH * 2);
  _Float16* wqkv_t  = (_Float16*)alloc((size_t)NQKV * CCH * 2);
  _Float16* wproj_t = (_Float16*)alloc((size_t)CCH * CCH * 2);
  _Float16* xT      = (_Float16*)alloc((size_t)BSZ * LP * CCH * 2);
  _Float16* xs      = (_Float16*)alloc((size_t)BSZ * LP * CCH * 2);
  const size_t seqElems = (size_t)BSZ * DILC * NHEAD * GN * DH;
  _Float16* qarr = (_Float16*)alloc(seqElems * 2);
  _Float16* karr = (_Float16*)alloc(seqElems * 2);
  _Float16* varr = (_Float16*)alloc(seqElems * 2);
  _Float16* oarr = (_Float16*)alloc((size_t)BSZ * LP * CCH * 2);

  float* dres = (float*)d_out;
  float* doutb = dres + (size_t)BSZ * CCH * LSEQ;

  // 1. weight prep
  k_conv_weight<<<CCH, 256, 0, stream>>>(conv_v, conv_g, wconv_t);
  k_transpose_w<<<(CCH * NQKV + 255) / 256, 256, 0, stream>>>(wqkv, wqkv_t, CCH, NQKV);
  k_transpose_w<<<(CCH * CCH + 255) / 256, 256, 0, stream>>>(wproj, wproj_t, CCH, CCH);

  // 2. activation transpose + pad
  k_transpose_x<<<dim3((LP + 31) / 32, CCH / 32, BSZ), 256, 0, stream>>>(x, xT);
  k_zero_pad<<<(BSZ * PADL * CCH) / 256, 256, 0, stream>>>(xs);

  // 3. dilated conv (implicit WMMA GEMM) + ELU
  k_conv_gemm<<<dim3(BSZ * (LSEQ / 64), CCH / 128), 128, 0, stream>>>(
      xT, wconv_t, conv_b, xs);

  // 4. QKV GEMM + head scatter
  k_qkv_gemm<<<dim3(MROWS / 64, NQKV / 128), 128, 0, stream>>>(
      xs, wqkv_t, bqkv, qarr, karr, varr);

  // 5. neighborhood attention
  k_attn<<<dim3(BSZ * DILC * NHEAD, (GN + 255) / 256), 256, 0, stream>>>(
      qarr, karr, varr, rpb, oarr);

  // 6. projection GEMM + crop/ELU/residual
  k_proj_gemm<<<dim3(MROWS / 64, CCH / 128), 128, 0, stream>>>(
      oarr, wproj_t, bproj, x, dres, doutb);
}
